// Attention_49254684950687
// MI455X (gfx1250) — matrix operand order repaired and checked by reading, not compiled, not run on hardware
//
#include <hip/hip_runtime.h>

typedef __bf16 bf16;
typedef __attribute__((ext_vector_type(16))) __bf16 v16bf;
typedef __attribute__((ext_vector_type(8)))  __bf16 v8bf;
typedef __attribute__((ext_vector_type(8)))  float  v8f;

#define WMMA_BF16(a, b, c) \
  __builtin_amdgcn_wmma_f32_16x16x32_bf16(false, (a), false, (b), (short)0, (c), false, false)

#define BB    16
#define NN    1025
#define DIMC  768
#define NH    12
#define HD    64
#define NPAD  1056            /* 33 * 32, padded key length          */
#define QPAD  1040            /* 65 * 16, padded query length        */
#define MTOT  (BB * NN)       /* 16400                               */
#define QSCALE 0.125f         /* 64^-0.5                             */

static __device__ __forceinline__ int imin(int a, int b) { return a < b ? a : b; }

// ---------------------------------------------------------------- fp32 -> bf16
__global__ void cvt_f32_bf16(const float* __restrict__ in, bf16* __restrict__ out, int n) {
  int i = blockIdx.x * blockDim.x + threadIdx.x;
  if (i < n) out[i] = (bf16)in[i];
}

// ------------------------------------------------- zero Vt pad columns [1025,1056)
__global__ void pad_vt(bf16* __restrict__ Vt) {
  const int padw = NPAD - NN;                 // 31
  const int total = BB * NH * HD * padw;
  int i = blockIdx.x * blockDim.x + threadIdx.x;
  if (i < total) {
    int r = i / padw;
    int c = NN + (i - r * padw);
    Vt[(size_t)r * NPAD + c] = (bf16)0.0f;
  }
}

// ----------------------------------------- precompute relative-position bias
// BiasB[h][k][q] (q innermost!) = table[ridx[q,k]*NH + h]; pad -> -1e30.
// q-innermost lets the attention S-phase fetch 8 consecutive query rows with
// one 16B load per lane. 26 MB bf16, L2-resident, reused by all B*H slices.
__global__ void bias_pre(const float* __restrict__ table, const int* __restrict__ ridx,
                         bf16* __restrict__ BiasB) {
  const int total = NH * NPAD * QPAD;
  int i = blockIdx.x * blockDim.x + threadIdx.x;
  if (i >= total) return;
  const int h = i / (NPAD * QPAD);
  const int rem = i - h * (NPAD * QPAD);
  const int k = rem / QPAD;
  const int q = rem - k * QPAD;
  float v = -1e30f;
  if (k < NN && q < NN) v = table[(size_t)ridx[(size_t)q * NN + k] * NH + h];
  BiasB[i] = (bf16)v;
}

// ---------------------------------------------------------------- QKV GEMM
// Y = X(bf16)[M,768] @ Wqkv(bf16)[2304,768]^T + bias, scattered to Q/K/Vt.
// One wave computes a 32(M) x 64(N) tile: 8 WMMA per 32-deep k-step.
__global__ void qkv_gemm(const bf16* __restrict__ X, const bf16* __restrict__ W,
                         const float* __restrict__ qb, const float* __restrict__ vb,
                         bf16* __restrict__ Q, bf16* __restrict__ K,
                         bf16* __restrict__ Vt) {
  const int lane = threadIdx.x;
  const int half = lane >> 4;
  const int l16  = lane & 15;
  const int m0 = blockIdx.x * 32;
  const int n0 = blockIdx.y * 64;

  const bf16* xr0 = X + (size_t)imin(m0 + l16,      MTOT - 1) * DIMC;
  const bf16* xr1 = X + (size_t)imin(m0 + 16 + l16, MTOT - 1) * DIMC;

  v8f acc[2][4] = {};
  for (int k0 = 0; k0 < DIMC; k0 += 32) {
    union { v16bf v; v8bf p[2]; } a0, a1;
    a0.p[0] = *(const v8bf*)(xr0 + k0 + half * 16);
    a0.p[1] = *(const v8bf*)(xr0 + k0 + half * 16 + 8);
    a1.p[0] = *(const v8bf*)(xr1 + k0 + half * 16);
    a1.p[1] = *(const v8bf*)(xr1 + k0 + half * 16 + 8);
    const int ks = k0 + half * 16;
    v16bf b0 = *(const v16bf*)(W + (size_t)(n0 +  0 + l16) * DIMC + ks);
    v16bf b1 = *(const v16bf*)(W + (size_t)(n0 + 16 + l16) * DIMC + ks);
    v16bf b2 = *(const v16bf*)(W + (size_t)(n0 + 32 + l16) * DIMC + ks);
    v16bf b3 = *(const v16bf*)(W + (size_t)(n0 + 48 + l16) * DIMC + ks);
    acc[0][0] = WMMA_BF16(a0.v, b0, acc[0][0]);
    acc[0][1] = WMMA_BF16(a0.v, b1, acc[0][1]);
    acc[0][2] = WMMA_BF16(a0.v, b2, acc[0][2]);
    acc[0][3] = WMMA_BF16(a0.v, b3, acc[0][3]);
    acc[1][0] = WMMA_BF16(a1.v, b0, acc[1][0]);
    acc[1][1] = WMMA_BF16(a1.v, b1, acc[1][1]);
    acc[1][2] = WMMA_BF16(a1.v, b2, acc[1][2]);
    acc[1][3] = WMMA_BF16(a1.v, b3, acc[1][3]);
  }

#pragma unroll
  for (int t = 0; t < 4; ++t) {
    const int j = n0 + t * 16 + l16;          // out column in [0,2304)
    const int which = j / DIMC;               // 0=q 1=k 2=v
    const int r = j - which * DIMC;           // [0,768)
    const int hh = r >> 6, d = r & 63;
    const float bias = (which == 0) ? qb[r] : (which == 2) ? vb[r] : 0.0f;
#pragma unroll
    for (int g = 0; g < 2; ++g) {
#pragma unroll
      for (int i = 0; i < 8; ++i) {
        const int m = m0 + g * 16 + i + half * 8;
        if (m < MTOT) {
          const int bidx = m / NN;
          const int n = m - bidx * NN;
          const float v = acc[g][t][i] + bias;
          const size_t bh = (size_t)(bidx * NH + hh);
          if (which == 0)      Q [(bh * NN + n) * HD + d]   = (bf16)(v * QSCALE);
          else if (which == 1) K [(bh * NN + n) * HD + d]   = (bf16)v;
          else                 Vt[(bh * HD + d) * NPAD + n] = (bf16)v;
        }
      }
    }
  }
}

// ---------------------------------------------------------------- attention
// One wave per (b, h, 16-query tile). S kept in LDS as bf16 (34 KB -> ~9 WGs
// per WGP). Row max in registers during S-phase; row sum fused into PV A-build.
__global__ void attn_kernel(const bf16* __restrict__ Q, const bf16* __restrict__ K,
                            const bf16* __restrict__ Vt,
                            const bf16* __restrict__ BiasB,
                            bf16* __restrict__ O) {
  extern __shared__ char smem[];
  bf16*  Sb   = (bf16*)smem;                       // [16][NPAD]
  float* rowM = (float*)(smem + 16 * NPAD * 2);    // [16]
  float* rowL = rowM + 16;                         // [16]

  const int lane = threadIdx.x;
  const int half = lane >> 4;
  const int l16  = lane & 15;
  const int qt = blockIdx.x, hh = blockIdx.y, b = blockIdx.z;
  const int bh = b * NH + hh;
  const int q0 = qt * 16;

  // init S columns [1024,1056) (tile kt=64 overwrites 1024..1039)
  {
    const int c = 1024 + lane;
#pragma unroll
    for (int r = 0; r < 16; ++r) Sb[r * NPAD + c] = (bf16)(-1e30f);
  }

  // Q tile -> two A fragments (d = 0..31, 32..63)
  const int qrow = imin(q0 + l16, NN - 1);
  const bf16* qbase = Q + ((size_t)bh * NN + qrow) * HD;
  union { v16bf v; v8bf p[2]; } aQ0, aQ1;
  aQ0.p[0] = *(const v8bf*)(qbase + half * 16);
  aQ0.p[1] = *(const v8bf*)(qbase + half * 16 + 8);
  aQ1.p[0] = *(const v8bf*)(qbase + 32 + half * 16);
  aQ1.p[1] = *(const v8bf*)(qbase + 32 + half * 16 + 8);

  // ---- S = Q K^T + bias, with per-row running max in registers
  float pmax[8];
#pragma unroll
  for (int i = 0; i < 8; ++i) pmax[i] = -1e30f;

  const bf16* bbase = BiasB + (size_t)hh * NPAD * QPAD + q0 + half * 8;
  for (int kt = 0; kt < 65; ++kt) {
    const int k0 = kt * 16;
    const int krow = imin(k0 + l16, NN - 1);
    const bf16* kbase = K + ((size_t)bh * NN + krow) * HD + half * 16;
    v16bf b0 = *(const v16bf*)(kbase);
    v16bf b1 = *(const v16bf*)(kbase + 32);
    v8f c = {};
    c = WMMA_BF16(aQ0.v, b0, c);
    c = WMMA_BF16(aQ1.v, b1, c);
    const int kcol = k0 + l16;
    const v8bf bv = *(const v8bf*)(bbase + (size_t)kcol * QPAD);  // 8 query rows
#pragma unroll
    for (int i = 0; i < 8; ++i) {
      const int m = i + half * 8;
      const float v = c[i] + (float)bv[i];
      Sb[m * NPAD + kcol] = (bf16)v;
      pmax[i] = fmaxf(pmax[i], v);
    }
  }

  // reduce max across the 16 lanes sharing each C-layout row; publish to LDS
#pragma unroll
  for (int i = 0; i < 8; ++i) {
    pmax[i] = fmaxf(pmax[i], __shfl_xor(pmax[i], 1));
    pmax[i] = fmaxf(pmax[i], __shfl_xor(pmax[i], 2));
    pmax[i] = fmaxf(pmax[i], __shfl_xor(pmax[i], 4));
    pmax[i] = fmaxf(pmax[i], __shfl_xor(pmax[i], 8));
  }
  if (l16 == 0) {
#pragma unroll
    for (int i = 0; i < 8; ++i) rowM[i + half * 8] = pmax[i];
  }
  __syncthreads();

  // ---- O = exp(S - m) @ V, row-sum fused into the A-fragment build.
  // each lane half takes 16 contiguous keys of its S row (k = 16*half + s, same order as the Vt B loads):
  // two ds_load_b128 per lane per 32-key chunk.
  v8f o0 = {}, o1 = {}, o2 = {}, o3 = {};
  float lsum = 0.0f;
  const float mrow = rowM[l16];
  const bf16* vtb = Vt + (size_t)bh * HD * NPAD + half * 16;
  for (int c32 = 0; c32 < 33; ++c32) {
    const int kc = c32 * 32;
    const bf16* srow = Sb + l16 * NPAD + kc + half * 16;
    const v8bf s0 = *(const v8bf*)(srow);
    const v8bf s1 = *(const v8bf*)(srow + 8);
    v16bf aP;
#pragma unroll
    for (int j = 0; j < 8; ++j) {
      const bf16 e0 = (bf16)__expf((float)s0[j] - mrow);
      const bf16 e1 = (bf16)__expf((float)s1[j] - mrow);
      aP[j]     = e0;
      aP[j + 8] = e1;
      lsum += (float)e0 + (float)e1;
    }
    o0 = WMMA_BF16(aP, *(const v16bf*)(vtb + (size_t)( 0 + l16) * NPAD + kc), o0);
    o1 = WMMA_BF16(aP, *(const v16bf*)(vtb + (size_t)(16 + l16) * NPAD + kc), o1);
    o2 = WMMA_BF16(aP, *(const v16bf*)(vtb + (size_t)(32 + l16) * NPAD + kc), o2);
    o3 = WMMA_BF16(aP, *(const v16bf*)(vtb + (size_t)(48 + l16) * NPAD + kc), o3);
  }
  lsum += __shfl_xor(lsum, 16);
  if (half == 0) rowL[l16] = lsum;
  __syncthreads();

  v8f oc[4] = {o0, o1, o2, o3};
#pragma unroll
  for (int g = 0; g < 4; ++g) {
#pragma unroll
    for (int i = 0; i < 8; ++i) {
      const int m = i + half * 8;
      const int qi = q0 + m;
      if (qi < NN) {
        const float inv = 1.0f / rowL[m];
        O[((size_t)b * NN + qi) * DIMC + hh * HD + g * 16 + l16] = (bf16)(oc[g][i] * inv);
      }
    }
  }
}

// ---------------------------------------------------------------- output proj
// out = A(bf16)[M,768] @ proj_w(bf16)[768,768]^T + proj_b, fp32 out.
__global__ void proj_gemm(const bf16* __restrict__ A, const bf16* __restrict__ W,
                          const float* __restrict__ bias, float* __restrict__ out) {
  const int lane = threadIdx.x;
  const int half = lane >> 4;
  const int l16  = lane & 15;
  const int m0 = blockIdx.x * 32;
  const int n0 = blockIdx.y * 64;

  const bf16* ar0 = A + (size_t)imin(m0 + l16,      MTOT - 1) * DIMC;
  const bf16* ar1 = A + (size_t)imin(m0 + 16 + l16, MTOT - 1) * DIMC;

  v8f acc[2][4] = {};
  for (int k0 = 0; k0 < DIMC; k0 += 32) {
    union { v16bf v; v8bf p[2]; } a0, a1;
    a0.p[0] = *(const v8bf*)(ar0 + k0 + half * 16);
    a0.p[1] = *(const v8bf*)(ar0 + k0 + half * 16 + 8);
    a1.p[0] = *(const v8bf*)(ar1 + k0 + half * 16);
    a1.p[1] = *(const v8bf*)(ar1 + k0 + half * 16 + 8);
    const int ks = k0 + half * 16;
    v16bf b0 = *(const v16bf*)(W + (size_t)(n0 +  0 + l16) * DIMC + ks);
    v16bf b1 = *(const v16bf*)(W + (size_t)(n0 + 16 + l16) * DIMC + ks);
    v16bf b2 = *(const v16bf*)(W + (size_t)(n0 + 32 + l16) * DIMC + ks);
    v16bf b3 = *(const v16bf*)(W + (size_t)(n0 + 48 + l16) * DIMC + ks);
    acc[0][0] = WMMA_BF16(a0.v, b0, acc[0][0]);
    acc[0][1] = WMMA_BF16(a0.v, b1, acc[0][1]);
    acc[0][2] = WMMA_BF16(a0.v, b2, acc[0][2]);
    acc[0][3] = WMMA_BF16(a0.v, b3, acc[0][3]);
    acc[1][0] = WMMA_BF16(a1.v, b0, acc[1][0]);
    acc[1][1] = WMMA_BF16(a1.v, b1, acc[1][1]);
    acc[1][2] = WMMA_BF16(a1.v, b2, acc[1][2]);
    acc[1][3] = WMMA_BF16(a1.v, b3, acc[1][3]);
  }

#pragma unroll
  for (int t = 0; t < 4; ++t) {
    const int col = n0 + t * 16 + l16;
    const float pb = bias[col];
#pragma unroll
    for (int g = 0; g < 2; ++g) {
#pragma unroll
      for (int i = 0; i < 8; ++i) {
        const int m = m0 + g * 16 + i + half * 8;
        if (m < MTOT) out[(size_t)m * DIMC + col] = acc[g][t][i] + pb;
      }
    }
  }
}

// ---------------------------------------------------------------- host launch
extern "C" void kernel_launch(void* const* d_in, const int* in_sizes, int n_in,
                              void* d_out, int out_size, void* d_ws, size_t ws_size,
                              hipStream_t stream) {
  (void)in_sizes; (void)n_in; (void)out_size; (void)ws_size;
  const float* x      = (const float*)d_in[0];
  const float* qkv_w  = (const float*)d_in[1];
  const float* q_bias = (const float*)d_in[2];
  const float* v_bias = (const float*)d_in[3];
  const float* table  = (const float*)d_in[4];
  const float* proj_w = (const float*)d_in[5];
  const float* proj_b = (const float*)d_in[6];
  const int*   ridx   = (const int*)d_in[7];
  float* out = (float*)d_out;

  char* ws = (char*)d_ws;
  size_t off = 0;
  auto alloc = [&](size_t bytes) -> void* {
    void* p = ws + off;
    off = (off + bytes + 255) & ~(size_t)255;
    return p;
  };
  bf16* Xb = (bf16*)alloc((size_t)MTOT * DIMC * 2);           // x in bf16
  bf16* Wq = (bf16*)alloc((size_t)3 * DIMC * DIMC * 2);       // qkv_w bf16
  bf16* Wp = (bf16*)alloc((size_t)DIMC * DIMC * 2);           // proj_w bf16
  bf16* Qb = (bf16*)alloc((size_t)BB * NH * NN * HD * 2);     // [B,H,N,hd] (pre-scaled)
  bf16* Kb = (bf16*)alloc((size_t)BB * NH * NN * HD * 2);     // [B,H,N,hd]
  bf16* Vt = (bf16*)alloc((size_t)BB * NH * HD * NPAD * 2);   // [B,H,hd,NPAD]
  bf16* Ob = (bf16*)alloc((size_t)MTOT * DIMC * 2);           // attention out [B,N,DIM]
  bf16* Bi = (bf16*)alloc((size_t)NH * NPAD * QPAD * 2);      // bias [H,NPAD,QPAD]

  const int nx = MTOT * DIMC;
  const int nwq = 3 * DIMC * DIMC;
  const int nwp = DIMC * DIMC;
  cvt_f32_bf16<<<(nx  + 255) / 256, 256, 0, stream>>>(x,      Xb, nx);
  cvt_f32_bf16<<<(nwq + 255) / 256, 256, 0, stream>>>(qkv_w,  Wq, nwq);
  cvt_f32_bf16<<<(nwp + 255) / 256, 256, 0, stream>>>(proj_w, Wp, nwp);

  const int npad = BB * NH * HD * (NPAD - NN);
  pad_vt<<<(npad + 255) / 256, 256, 0, stream>>>(Vt);

  const int nbias = NH * NPAD * QPAD;
  bias_pre<<<(nbias + 255) / 256, 256, 0, stream>>>(table, ridx, Bi);

  qkv_gemm<<<dim3((MTOT + 31) / 32, (3 * DIMC) / 64), 32, 0, stream>>>(
      Xb, Wq, q_bias, v_bias, Qb, Kb, Vt);

  const size_t smem = (size_t)(16 * NPAD * 2) + 32 * sizeof(float);
  attn_kernel<<<dim3(65, NH, BB), 32, smem, stream>>>(Qb, Kb, Vt, Bi, Ob);

  proj_gemm<<<dim3((MTOT + 31) / 32, DIMC / 64), 32, 0, stream>>>(Ob, Wp, proj_b, out);
}
